// MHA_70239895159407
// MI455X (gfx1250) — matrix operand order repaired and checked by reading, not compiled, not run on hardware
//
#include <hip/hip_runtime.h>
#include <hip/hip_bf16.h>
#include <math.h>

typedef __bf16 bf16;
typedef __attribute__((ext_vector_type(16))) __bf16 v16bf;
typedef __attribute__((ext_vector_type(8)))  __bf16 v8bf;
typedef __attribute__((ext_vector_type(4)))  __bf16 v4bf;
typedef __attribute__((ext_vector_type(8)))  float  v8f;

#define BB  2
#define TT  2048
#define DD  1024
#define HH  16
#define KHH 64

// ---------------------------------------------------------------------------
// WMMA helpers (CDNA5 wave32, 16x16x32 bf16 -> f32)
// ---------------------------------------------------------------------------
static __device__ __forceinline__ v8f wmma_bf16(v16bf a, v16bf b, v8f c) {
  return __builtin_amdgcn_wmma_f32_16x16x32_bf16(
      /*neg_a=*/false, a, /*neg_b=*/false, b,
      /*c_mod=*/(short)0, c, /*reuse_a=*/false, /*reuse_b=*/false);
}

// A-matrix 16x32 (MxK), row-major source with leading dim ld.
// Lanes 0-15 hold rows M=0..15 with K={0..7, 16..23}; lanes 16-31 same rows,
// K={8..15, 24..31}.  (ISA 7.12.2)
static __device__ __forceinline__ v16bf load_A_frag(const bf16* __restrict__ base,
                                                    int ld, int k0) {
  int lane = threadIdx.x & 31;
  int m = lane & 15, hi = lane >> 4;
  const bf16* p = base + (size_t)m * ld + k0 + 8 * hi;
  v8bf a0 = *(const v8bf*)p;
  v8bf a1 = *(const v8bf*)(p + 16);
  v16bf r;
#pragma unroll
  for (int i = 0; i < 8; ++i) { r[i] = a0[i]; r[8 + i] = a1[i]; }
  return r;
}

// B-matrix 32x16 (KxN): element (k,n) taken from src[n][k] (row-major rows=n).
// Lanes 0-15: column N=lane, K={0..7, 16..23}; lanes 16-31: K={8..15, 24..31}.
static __device__ __forceinline__ v16bf load_B_frag(const bf16* __restrict__ base,
                                                    int ld, int k0) {
  int lane = threadIdx.x & 31;
  int n = lane & 15, hi = lane >> 4;
  const bf16* p = base + (size_t)n * ld + k0 + 8 * hi;
  v8bf b0 = *(const v8bf*)p;
  v8bf b1 = *(const v8bf*)(p + 16);
  v16bf r;
#pragma unroll
  for (int i = 0; i < 8; ++i) { r[i] = b0[i]; r[8 + i] = b1[i]; }
  return r;
}

// Speculative prefetch of the fragment a wave will need next iteration
// (emits global_prefetch_b8; one line per lane row).
static __device__ __forceinline__ void prefetch_B_rows(const bf16* base, int ld,
                                                       int k0) {
  int lane = threadIdx.x & 31;
  int n = lane & 15, hi = lane >> 4;
  __builtin_prefetch(base + (size_t)n * ld + k0 + 16 * hi, 0, 3);
}

// ---------------------------------------------------------------------------
// fp32 -> bf16 conversion (vectorized x4)
// ---------------------------------------------------------------------------
__global__ void cvt_kernel(const float* __restrict__ in, bf16* __restrict__ out,
                           int n4) {
  int i = blockIdx.x * blockDim.x + threadIdx.x;
  if (i < n4) {
    float4 v = ((const float4*)in)[i];
    v4bf o;
    o[0] = (bf16)v.x; o[1] = (bf16)v.y; o[2] = (bf16)v.z; o[3] = (bf16)v.w;
    ((v4bf*)out)[i] = o;
  }
}

// ---------------------------------------------------------------------------
// y = A(MxK) * W(NxK)^T GEMM; one wave computes a 32x64 output tile
// (2 A-fragments amortize each B-fragment -> ~21 flops/byte from L2).
// MODE 0: store bf16, head-split [B,H,T,64]          (Q projection)
// MODE 1: cols<1024 -> head-split K;  cols>=1024 -> V transposed [B,H,64,T]
// MODE 2: store f32 row-major [M,N]                  (output projection)
// ---------------------------------------------------------------------------
template <int MODE>
__global__ void __launch_bounds__(256)
gemm_kernel(const bf16* __restrict__ A, const bf16* __restrict__ W,
            bf16* __restrict__ outB, bf16* __restrict__ outVT,
            float* __restrict__ outF, int M, int N, int K) {
  int wid = blockIdx.x * (blockDim.x >> 5) + (threadIdx.x >> 5);
  int tN = N >> 6;
  int im = wid / tN, in4 = wid % tN;
  int m0 = im << 5, n0 = in4 << 6;
  if (m0 >= M) return;

  v8f acc[2][4] = {};
  const bf16* A0 = A + (size_t)m0 * K;
  const bf16* A1 = A + (size_t)(m0 + 16) * K;
  for (int k0 = 0; k0 < K; k0 += 32) {
    v16bf a0 = load_A_frag(A0, K, k0);
    v16bf a1 = load_A_frag(A1, K, k0);
#pragma unroll
    for (int j = 0; j < 4; ++j) {
      const bf16* Wj = W + (size_t)(n0 + 16 * j) * K;
      v16bf b = load_B_frag(Wj, K, k0);
      acc[0][j] = wmma_bf16(a0, b, acc[0][j]);
      acc[1][j] = wmma_bf16(a1, b, acc[1][j]);
      prefetch_B_rows(Wj, K, k0 + 32);      // next k-step into WGP$/L2
    }
  }

  int lane = threadIdx.x & 31;
  int nl = lane & 15, hi = lane >> 4;
#pragma unroll
  for (int half = 0; half < 2; ++half) {
#pragma unroll
    for (int j = 0; j < 4; ++j) {
#pragma unroll
      for (int r = 0; r < 8; ++r) {
        int row = m0 + 16 * half + r + 8 * hi;   // token index over B*T
        int col = n0 + 16 * j + nl;
        float v = acc[half][j][r];
        if (MODE == 2) {
          outF[(size_t)row * N + col] = v;
        } else {
          int b = row / TT, t = row % TT;
          if (MODE == 0 || col < DD) {
            int h = col >> 6, kh = col & 63;
            outB[((size_t)(b * HH + h) * TT + t) * KHH + kh] = (bf16)v;
          } else {
            int c = col - DD;
            int h = c >> 6, kh = c & 63;
            outVT[((size_t)(b * HH + h) * KHH + kh) * TT + t] = (bf16)v;
          }
        }
      }
    }
  }
}

// ---------------------------------------------------------------------------
// Flash attention: one wave = 16 query rows of one (b,h); 64 keys per step.
// Row-sums come from a WMMA against an all-ones B fragment (same recurrence
// as O), so only the row-max needs a cross-lane butterfly.
// Exactly one tile per wave is partially masked (MASKED=true epilogue).
// ---------------------------------------------------------------------------
template <bool MASKED>
static __device__ __forceinline__ void fa_tile(
    int kt, int q0, int nl, int hi, const bf16* __restrict__ Kh,
    const bf16* __restrict__ Vh, bf16* sm, v16bf aQ0, v16bf aQ1, v16bf ones,
    float (&mrow)[8], v8f& lacc, v8f (&oacc)[4]) {
  const float sc = 0.125f;  // 1/sqrt(64)
  // ---- S = Q K^T : four 16x16 tiles over 64 keys ----
  v8f s[4] = {};
#pragma unroll
  for (int sub = 0; sub < 4; ++sub) {
    if (!MASKED || (kt + 16 * sub) <= q0 + 15) {
      v16bf b0 = load_B_frag(Kh + (size_t)(kt + 16 * sub) * KHH, KHH, 0);
      s[sub] = wmma_bf16(aQ0, b0, s[sub]);
      v16bf b1 = load_B_frag(Kh + (size_t)(kt + 16 * sub) * KHH, KHH, 32);
      s[sub] = wmma_bf16(aQ1, b1, s[sub]);
      if (!MASKED)   // pull next tile's K rows toward WGP$ while WMMAs run
        prefetch_B_rows(Kh + (size_t)(kt + 64 + 16 * sub) * KHH, KHH, 0);
    }
  }
  // ---- scale (+ causal mask only in the diagonal tile) ----
#pragma unroll
  for (int sub = 0; sub < 4; ++sub)
#pragma unroll
    for (int r = 0; r < 8; ++r) {
      float v = s[sub][r] * sc;
      if (MASKED) {
        int kcol = kt + 16 * sub + nl;
        int qrow = q0 + r + 8 * hi;
        v = (kcol <= qrow) ? v : -3.0e38f;
      }
      s[sub][r] = v;
    }
  // ---- row max (only remaining cross-lane reduction) ----
  float tm[8];
#pragma unroll
  for (int r = 0; r < 8; ++r)
    tm[r] = fmaxf(fmaxf(s[0][r], s[1][r]), fmaxf(s[2][r], s[3][r]));
#pragma unroll
  for (int d = 1; d < 16; d <<= 1)
#pragma unroll
    for (int r = 0; r < 8; ++r) tm[r] = fmaxf(tm[r], __shfl_xor(tm[r], d, 32));

  float alpha[8];
#pragma unroll
  for (int r = 0; r < 8; ++r) {
    float mn = fmaxf(mrow[r], tm[r]);
    alpha[r] = __expf(mrow[r] - mn);
    mrow[r] = mn;
  }
#pragma unroll
  for (int sub = 0; sub < 4; ++sub)
#pragma unroll
    for (int r = 0; r < 8; ++r) s[sub][r] = __expf(s[sub][r] - mrow[r]);

  // ---- rescale running accumulators (l and O share the recurrence) ----
#pragma unroll
  for (int r = 0; r < 8; ++r) lacc[r] *= alpha[r];
#pragma unroll
  for (int j = 0; j < 4; ++j)
#pragma unroll
    for (int r = 0; r < 8; ++r) oacc[j][r] *= alpha[r];

  // ---- P (C-layout) -> LDS row-major 16x64 -> A fragments ----
#pragma unroll
  for (int sub = 0; sub < 4; ++sub)
#pragma unroll
    for (int r = 0; r < 8; ++r)
      sm[(r + 8 * hi) * 64 + 16 * sub + nl] = (bf16)s[sub][r];

  v16bf aP0 = load_A_frag(sm, 64, 0);
  lacc = wmma_bf16(aP0, ones, lacc);          // row-sums via matrix pipe
#pragma unroll
  for (int j = 0; j < 4; ++j) {
    v16bf bV = load_B_frag(Vh + (size_t)(16 * j) * TT, TT, kt);
    oacc[j] = wmma_bf16(aP0, bV, oacc[j]);
  }
  // Upper 32 keys of the tile: skip entirely if diagonal doesn't reach them.
  if (!MASKED || (q0 - kt) >= 32) {
    v16bf aP1 = load_A_frag(sm, 64, 32);
    lacc = wmma_bf16(aP1, ones, lacc);
#pragma unroll
    for (int j = 0; j < 4; ++j) {
      v16bf bV = load_B_frag(Vh + (size_t)(16 * j) * TT, TT, kt + 32);
      oacc[j] = wmma_bf16(aP1, bV, oacc[j]);
    }
  }
}

__global__ void __launch_bounds__(256)
fa_kernel(const bf16* __restrict__ Q, const bf16* __restrict__ Km,
          const bf16* __restrict__ VT, bf16* __restrict__ O) {
  __shared__ bf16 smem[8 * 16 * 64];          // 2 KB per wave
  int wslot = threadIdx.x >> 5;
  bf16* sm = &smem[wslot * (16 * 64)];

  int wid = blockIdx.x * 8 + wslot;
  const int nQ = TT / 16;                      // 128 query tiles per head
  int bh = wid / nQ;
  // Reverse mapping: heaviest (largest q0) tiles launch first -> better tail.
  int iq = (nQ - 1) - (wid % nQ);
  int b = bh / HH, h = bh % HH;
  int q0 = iq << 4;

  const bf16* Qh = Q  + (size_t)bh * TT * KHH;
  const bf16* Kh = Km + (size_t)bh * TT * KHH;
  const bf16* Vh = VT + (size_t)bh * KHH * TT;

  int lane = threadIdx.x & 31;
  int nl = lane & 15, hi = lane >> 4;

  v16bf aQ0 = load_A_frag(Qh + (size_t)q0 * KHH, KHH, 0);
  v16bf aQ1 = load_A_frag(Qh + (size_t)q0 * KHH, KHH, 32);
  v16bf ones;
#pragma unroll
  for (int i = 0; i < 16; ++i) ones[i] = (bf16)1.0f;

  float mrow[8];
#pragma unroll
  for (int r = 0; r < 8; ++r) mrow[r] = -3.0e38f;
  v8f lacc = {};
  v8f oacc[4] = {};

  int kt0 = q0 & ~63;                          // base of the diagonal tile
  for (int kt = 0; kt < kt0; kt += 64)
    fa_tile<false>(kt, q0, nl, hi, Kh, Vh, sm, aQ0, aQ1, ones, mrow, lacc, oacc);
  fa_tile<true>(kt0, q0, nl, hi, Kh, Vh, sm, aQ0, aQ1, ones, mrow, lacc, oacc);

  float linv[8];
#pragma unroll
  for (int r = 0; r < 8; ++r) linv[r] = 1.f / lacc[r];
#pragma unroll
  for (int j = 0; j < 4; ++j)
#pragma unroll
    for (int r = 0; r < 8; ++r) {
      int t = q0 + r + 8 * hi;
      O[(size_t)(b * TT + t) * DD + h * KHH + 16 * j + nl] =
          (bf16)(oacc[j][r] * linv[r]);
    }
}

// ---------------------------------------------------------------------------
// Host launcher
// ---------------------------------------------------------------------------
extern "C" void kernel_launch(void* const* d_in, const int* in_sizes, int n_in,
                              void* d_out, int out_size, void* d_ws, size_t ws_size,
                              hipStream_t stream) {
  (void)in_sizes; (void)n_in; (void)out_size; (void)ws_size;
  const float* x    = (const float*)d_in[0];
  const float* Wq   = (const float*)d_in[1];
  const float* Wkv  = (const float*)d_in[2];
  const float* Wout = (const float*)d_in[3];
  float* out = (float*)d_out;

  char* ws = (char*)d_ws;
  size_t off = 0;
  auto walloc = [&](size_t bytes) -> char* {
    char* p = ws + off;
    off += (bytes + 255) & ~(size_t)255;
    return p;
  };
  const size_t M = (size_t)BB * TT;                 // 4096 tokens
  bf16* xb   = (bf16*)walloc(M * DD * 2);           // 8 MB
  bf16* wqb  = (bf16*)walloc((size_t)DD * DD * 2);  // 2 MB
  bf16* wkvb = (bf16*)walloc((size_t)2 * DD * DD * 2); // 4 MB
  bf16* wob  = (bf16*)walloc((size_t)DD * DD * 2);  // 2 MB
  bf16* Qb   = (bf16*)walloc(M * DD * 2);           // 8 MB  [B,H,T,64]
  bf16* Kb   = (bf16*)walloc(M * DD * 2);           // 8 MB  [B,H,T,64]
  bf16* VTb  = (bf16*)walloc(M * DD * 2);           // 8 MB  [B,H,64,T]
  bf16* Ob   = (bf16*)walloc(M * DD * 2);           // 8 MB  [B,T,1024]

  auto cvt = [&](const float* in, bf16* o, size_t n) {
    int n4 = (int)(n / 4);
    cvt_kernel<<<(n4 + 255) / 256, 256, 0, stream>>>(in, o, n4);
  };
  cvt(x,    xb,   M * DD);
  cvt(Wq,   wqb,  (size_t)DD * DD);
  cvt(Wkv,  wkvb, (size_t)2 * DD * DD);
  cvt(Wout, wob,  (size_t)DD * DD);

  // Q projection:  [M,1024] = xb * Wq^T
  {
    int tiles = ((int)M / 32) * (DD / 64);          // 2048 waves
    gemm_kernel<0><<<tiles / 8, 256, 0, stream>>>(xb, wqb, Qb, nullptr, nullptr,
                                                  (int)M, DD, DD);
  }
  // KV projection: [M,2048] = xb * Wkv^T  (K head-split, V transposed)
  {
    int tiles = ((int)M / 32) * ((2 * DD) / 64);    // 4096 waves
    gemm_kernel<1><<<tiles / 8, 256, 0, stream>>>(xb, wkvb, Kb, VTb, nullptr,
                                                  (int)M, 2 * DD, DD);
  }
  // Flash attention
  {
    int waves = BB * HH * (TT / 16);                // 4096 waves
    fa_kernel<<<waves / 8, 256, 0, stream>>>(Qb, Kb, VTb, Ob);
  }
  // Output projection: out[M,1024] = Ob * Wout^T  (f32)
  {
    int tiles = ((int)M / 32) * (DD / 64);
    gemm_kernel<2><<<tiles / 8, 256, 0, stream>>>(Ob, wob, nullptr, nullptr, out,
                                                  (int)M, DD, DD);
  }
}
